// Net_22514218566151
// MI455X (gfx1250) — matrix operand order repaired and checked by reading, not compiled, not run on hardware
//
#include <hip/hip_runtime.h>
#include <hip/hip_bf16.h>
#include <math.h>

typedef __attribute__((ext_vector_type(16))) __bf16          v16bf;
typedef __attribute__((ext_vector_type(16))) unsigned short  v16u;
typedef __attribute__((ext_vector_type(8)))  unsigned short  v8us;
typedef __attribute__((ext_vector_type(8)))  float           v8f;

#define EPS 1e-16f
#define NEG_SLOPE 0.2f

// ---------- helpers ----------
__device__ __forceinline__ unsigned short f2bf(float f) {
  unsigned u = __float_as_uint(f);
  u += 0x7FFFu + ((u >> 16) & 1u);          // round-to-nearest-even
  return (unsigned short)(u >> 16);
}
__device__ __forceinline__ unsigned pack2bf(float a, float b) {
  unsigned ua = __float_as_uint(a); ua += 0x7FFFu + ((ua >> 16) & 1u);
  unsigned ub = __float_as_uint(b); ub += 0x7FFFu + ((ub >> 16) & 1u);
  return (ua >> 16) | (ub & 0xFFFF0000u);
}
// monotone float -> uint mapping so atomicMax(uint) orders like float
__device__ __forceinline__ unsigned fmap(float f) {
  unsigned u = __float_as_uint(f);
  return (u & 0x80000000u) ? ~u : (u | 0x80000000u);
}
__device__ __forceinline__ float funmap(unsigned u) {
  return (u & 0x80000000u) ? __uint_as_float(u ^ 0x80000000u)
                           : __uint_as_float(~u);
}
__device__ __forceinline__ float leaky(float g) {
  return g > 0.0f ? g : NEG_SLOPE * g;
}

// ---------- pre-pass: X (f32) -> Xb (bf16, row-major) ----------
__global__ void cvt_x_bf16(const float* __restrict__ X,
                           unsigned* __restrict__ Xb, int n2) {
  int t = blockIdx.x * blockDim.x + threadIdx.x;   // one packed pair per thread
  if (t >= n2) return;
  Xb[t] = pack2bf(X[2 * t], X[2 * t + 1]);
}

// ---------- pre-pass: W1 (f32 [512,64]) -> WMMA-B fragment order ----------
// layout: [chunk c:16][colgroup g:4][lane l:32][elem i:16]; lane elem i is
// B[k = 32c + 16*(l>>4) + i][col = 16g + (l&15)]
__global__ void swizzle_w_bf16(const float* __restrict__ W,
                               unsigned short* __restrict__ Wsw, int NC) {
  int t = blockIdx.x * blockDim.x + threadIdx.x;
  if (t >= 32768) return;
  int i = t & 15, l = (t >> 4) & 31, g = (t >> 9) & 3, c = t >> 11;
  int k   = 32 * c + 16 * (l >> 4) + i;
  int col = 16 * g + (l & 15);
  Wsw[t] = f2bf(W[k * NC + col]);
}

// ---------- GEMM (fast path): h1 = Xb @ Wsw via bf16 WMMA ----------
// grid.x = N/16, block = 128 (4 waves; wave w owns cols 16w..16w+15)
__global__ void gemm1_wmma_bf16(const unsigned short* __restrict__ Xb,
                                const unsigned short* __restrict__ Wsw,
                                float* __restrict__ H, int K, int NC) {
  const int m0   = blockIdx.x * 16;
  const int wave = threadIdx.x >> 5;
  const int lane = threadIdx.x & 31;
  const int half = lane >> 4;

  const unsigned short* xr = Xb + (size_t)(m0 + (lane & 15)) * K;
  const unsigned short* wr = Wsw + ((size_t)wave * 32 + lane) * 16;
  v8f c = {};

#pragma unroll 4
  for (int k0 = 0; k0 < K; k0 += 32) {
    v8us alo = *(const v8us*)(xr + k0 + 16 * half);       // K = k0+16h .. +7
    v8us ahi = *(const v8us*)(xr + k0 + 16 * half + 8);   // K = k0+16h+8 .. +7
    v8us blo = *(const v8us*)(wr);
    v8us bhi = *(const v8us*)(wr + 8);
    wr += 4 * 32 * 16;                                    // next K-chunk
    v16u au = __builtin_shufflevector(alo, ahi, 0, 1, 2, 3, 4, 5, 6, 7,
                                      8, 9, 10, 11, 12, 13, 14, 15);
    v16u bu = __builtin_shufflevector(blo, bhi, 0, 1, 2, 3, 4, 5, 6, 7,
                                      8, 9, 10, 11, 12, 13, 14, 15);
    v16bf a = __builtin_bit_cast(v16bf, au);
    v16bf b = __builtin_bit_cast(v16bf, bu);
    c = __builtin_amdgcn_wmma_f32_16x16x32_bf16(false, a, false, b,
                                                (short)0, c, false, false);
  }

  const int n     = lane & 15;
  const int rbase = (lane < 16) ? 0 : 8;
  const int n0    = wave * 16;
#pragma unroll
  for (int r = 0; r < 8; ++r)
    H[(size_t)(m0 + rbase + r) * NC + n0 + n] = c[r];
}

// ---------- GEMM (fallback, f32 source, in-register conversion) ----------
__global__ void gemm1_wmma_f32(const float* __restrict__ X,
                               const float* __restrict__ W,
                               float* __restrict__ H, int K, int NC) {
  const int m0   = blockIdx.x * 16;
  const int wave = threadIdx.x >> 5;
  const int lane = threadIdx.x & 31;
  const int n0   = wave * 16;
  const int half = lane >> 4;

  const float* xrow = X + (size_t)(m0 + (lane & 15)) * K;
  v8f c = {};

  for (int k0 = 0; k0 < K; k0 += 32) {
    v16u au, bu;
#pragma unroll
    for (int i = 0; i < 8; ++i) au[i]     = f2bf(xrow[k0 + 16 * half + i]);
#pragma unroll
    for (int i = 0; i < 8; ++i) au[8 + i] = f2bf(xrow[k0 + 16 * half + 8 + i]);
#pragma unroll
    for (int i = 0; i < 16; ++i)
      bu[i] = f2bf(W[(size_t)(k0 + 16 * half + i) * NC + n0 + (lane & 15)]);
    v16bf a = __builtin_bit_cast(v16bf, au);
    v16bf b = __builtin_bit_cast(v16bf, bu);
    c = __builtin_amdgcn_wmma_f32_16x16x32_bf16(false, a, false, b,
                                                (short)0, c, false, false);
  }
  const int n     = lane & 15;
  const int rbase = (lane < 16) ? 0 : 8;
#pragma unroll
  for (int r = 0; r < 8; ++r)
    H[(size_t)(m0 + rbase + r) * NC + n0 + n] = c[r];
}

// ---------- per-node attention terms + init stats (layer 1) ----------
__global__ void node1_prep(const float* __restrict__ h1,
                           const float* __restrict__ a_src,
                           const float* __restrict__ a_dst,
                           float* __restrict__ al, float* __restrict__ ar,
                           unsigned* __restrict__ mu, float* __restrict__ den,
                           float* __restrict__ outraw, int N) {
  int n = blockIdx.x * blockDim.x + threadIdx.x;
  if (n >= N) return;
  const float* hr = h1 + (size_t)n * 64;
#pragma unroll
  for (int h = 0; h < 8; ++h) {
    float sa = 0.f, sb = 0.f;
#pragma unroll
    for (int cc = 0; cc < 8; ++cc) {
      float v = hr[h * 8 + cc];
      sa += v * a_src[h * 8 + cc];
      sb += v * a_dst[h * 8 + cc];
    }
    al[n * 8 + h] = sa;
    ar[n * 8 + h] = sb;
    mu[n * 8 + h] = 0u;
    den[n * 8 + h] = 0.f;
  }
  float* o = outraw + (size_t)n * 64;
#pragma unroll
  for (int k = 0; k < 64; ++k) o[k] = 0.f;
}

// ---------- edge pass A (segment max), layer 1 ----------
__global__ void edge1_max(const long long* __restrict__ ei, int E, int ET,
                          const float* __restrict__ al, const float* __restrict__ ar,
                          unsigned* __restrict__ mu) {
  int t = blockIdx.x * blockDim.x + threadIdx.x;
  if (t >= ET * 8) return;
  int e = t >> 3, h = t & 7;
  int s, d;
  if (e < E) { s = (int)ei[e]; d = (int)ei[E + e]; }
  else       { s = d = e - E; }
  float g = leaky(al[s * 8 + h] + ar[d * 8 + h]);
  atomicMax(&mu[d * 8 + h], fmap(g));
}

// ---------- edge pass B (exp, denom, unnormalized messages), layer 1 ----------
__global__ void edge1_acc(const long long* __restrict__ ei, int E, int ET,
                          const float* __restrict__ al, const float* __restrict__ ar,
                          const unsigned* __restrict__ mu,
                          const float* __restrict__ h1,
                          float* __restrict__ den, float* __restrict__ outraw) {
  int t = blockIdx.x * blockDim.x + threadIdx.x;
  if (t >= ET * 8) return;
  int e = t >> 3, h = t & 7;
  int s, d;
  if (e < E) { s = (int)ei[e]; d = (int)ei[E + e]; }
  else       { s = d = e - E; }
  float g  = leaky(al[s * 8 + h] + ar[d * 8 + h]);
  float ex = __expf(g - funmap(mu[d * 8 + h]));
  atomicAdd(&den[d * 8 + h], ex);
  const float* hs = h1 + (size_t)s * 64 + h * 8;
  float* od = outraw + (size_t)d * 64 + h * 8;
#pragma unroll
  for (int cc = 0; cc < 8; ++cc) atomicAdd(&od[cc], ex * hs[cc]);
}

// ---------- normalize + bias + ELU (in place), layer 1 ----------
__global__ void node1_finish(float* __restrict__ outraw,
                             const float* __restrict__ den,
                             const float* __restrict__ b1, int N) {
  int t = blockIdx.x * blockDim.x + threadIdx.x;
  if (t >= N * 64) return;
  int n = t >> 6, k = t & 63, h = k >> 3;
  float v = outraw[t] / (den[n * 8 + h] + EPS) + b1[k];
  outraw[t] = v > 0.f ? v : (__expf(v) - 1.f);   // ELU
}

// ---------- layer-2 projection + attention terms + init ----------
__global__ void node2_proj(const float* __restrict__ hb,
                           const float* __restrict__ W2,
                           const float* __restrict__ a_src,
                           const float* __restrict__ a_dst,
                           float* __restrict__ h2, float* __restrict__ al,
                           float* __restrict__ ar, unsigned* __restrict__ mu,
                           float* __restrict__ den, float* __restrict__ outraw,
                           int N) {
  int n = blockIdx.x * blockDim.x + threadIdx.x;
  if (n >= N) return;
  const float* hr = hb + (size_t)n * 64;
  float acc[10];
#pragma unroll
  for (int cc = 0; cc < 10; ++cc) acc[cc] = 0.f;
  for (int k = 0; k < 64; ++k) {
    float xv = hr[k];
    const float* wrow = W2 + k * 10;
#pragma unroll
    for (int cc = 0; cc < 10; ++cc) acc[cc] += xv * wrow[cc];
  }
  float sa = 0.f, sb = 0.f;
#pragma unroll
  for (int cc = 0; cc < 10; ++cc) {
    h2[(size_t)n * 10 + cc] = acc[cc];
    outraw[(size_t)n * 10 + cc] = 0.f;
    sa += acc[cc] * a_src[cc];
    sb += acc[cc] * a_dst[cc];
  }
  al[n] = sa; ar[n] = sb; mu[n] = 0u; den[n] = 0.f;
}

// ---------- edge pass A, layer 2 ----------
__global__ void edge2_max(const long long* __restrict__ ei, int E, int ET,
                          const float* __restrict__ al, const float* __restrict__ ar,
                          unsigned* __restrict__ mu) {
  int e = blockIdx.x * blockDim.x + threadIdx.x;
  if (e >= ET) return;
  int s, d;
  if (e < E) { s = (int)ei[e]; d = (int)ei[E + e]; }
  else       { s = d = e - E; }
  atomicMax(&mu[d], fmap(leaky(al[s] + ar[d])));
}

// ---------- edge pass B, layer 2 ----------
__global__ void edge2_acc(const long long* __restrict__ ei, int E, int ET,
                          const float* __restrict__ al, const float* __restrict__ ar,
                          const unsigned* __restrict__ mu,
                          const float* __restrict__ h2,
                          float* __restrict__ den, float* __restrict__ outraw) {
  int e = blockIdx.x * blockDim.x + threadIdx.x;
  if (e >= ET) return;
  int s, d;
  if (e < E) { s = (int)ei[e]; d = (int)ei[E + e]; }
  else       { s = d = e - E; }
  float ex = __expf(leaky(al[s] + ar[d]) - funmap(mu[d]));
  atomicAdd(&den[d], ex);
  const float* hs = h2 + (size_t)s * 10;
  float* od = outraw + (size_t)d * 10;
#pragma unroll
  for (int cc = 0; cc < 10; ++cc) atomicAdd(&od[cc], ex * hs[cc]);
}

// ---------- normalize + bias + log_softmax ----------
__global__ void node2_finish(const float* __restrict__ outraw,
                             const float* __restrict__ den,
                             const float* __restrict__ b2,
                             float* __restrict__ out, int N) {
  int n = blockIdx.x * blockDim.x + threadIdx.x;
  if (n >= N) return;
  float o[10];
  float inv = 1.0f / (den[n] + EPS);
  float mx = -1e30f;
#pragma unroll
  for (int cc = 0; cc < 10; ++cc) {
    o[cc] = outraw[(size_t)n * 10 + cc] * inv + b2[cc];
    mx = fmaxf(mx, o[cc]);
  }
  float s = 0.f;
#pragma unroll
  for (int cc = 0; cc < 10; ++cc) s += __expf(o[cc] - mx);
  float lse = mx + __logf(s);
#pragma unroll
  for (int cc = 0; cc < 10; ++cc) out[(size_t)n * 10 + cc] = o[cc] - lse;
}

extern "C" void kernel_launch(void* const* d_in, const int* in_sizes, int n_in,
                              void* d_out, int out_size, void* d_ws, size_t ws_size,
                              hipStream_t stream) {
  const float*     x      = (const float*)d_in[0];       // [N,512]
  const long long* ei     = (const long long*)d_in[1];   // [2,E] int64
  const float*     W1     = (const float*)d_in[2];       // [512,64]
  const float*     a_src1 = (const float*)d_in[3];       // [8,8]
  const float*     a_dst1 = (const float*)d_in[4];       // [8,8]
  const float*     b1     = (const float*)d_in[5];       // [64]
  const float*     W2     = (const float*)d_in[6];       // [64,10]
  const float*     a_src2 = (const float*)d_in[7];       // [10]
  const float*     a_dst2 = (const float*)d_in[8];       // [10]
  const float*     b2     = (const float*)d_in[9];       // [10]
  float* out = (float*)d_out;

  const int N  = in_sizes[0] / 512;      // 100000
  const int E  = in_sizes[1] / 2;        // 1600000
  const int ET = E + N;                  // + self loops

  // workspace carve-up (f32/u32 region first; bf16 region appended)
  float*    h1   = (float*)d_ws;                     // N*64
  float*    al1  = h1  + (size_t)N * 64;             // N*8
  float*    ar1  = al1 + (size_t)N * 8;              // N*8
  unsigned* m1u  = (unsigned*)(ar1 + (size_t)N * 8); // N*8
  float*    den1 = (float*)(m1u + (size_t)N * 8);    // N*8
  float*    out1 = den1 + (size_t)N * 8;             // N*64 (becomes ELU'd h)
  float*    h2   = out1 + (size_t)N * 64;            // N*10
  float*    al2  = h2  + (size_t)N * 10;             // N
  float*    ar2  = al2 + (size_t)N;                  // N
  unsigned* m2u  = (unsigned*)(ar2 + (size_t)N);     // N
  float*    den2 = (float*)(m2u + (size_t)N);        // N
  float*    out2 = den2 + (size_t)N;                 // N*10

  const size_t baseBytes = (size_t)N * 184 * sizeof(float);
  unsigned short* Xb  = (unsigned short*)((char*)d_ws + baseBytes); // N*512 bf16
  unsigned short* Wsw = Xb + (size_t)N * 512;                       // 32768 bf16
  const bool fast = ws_size >= baseBytes + (size_t)N * 512 * 2 + 65536;

  const int B = 256;

  // Layer 1 projection
  if (fast) {
    const int n2 = N * 256;                               // packed bf16 pairs
    cvt_x_bf16<<<(n2 + B - 1) / B, B, 0, stream>>>(x, (unsigned*)Xb, n2);
    swizzle_w_bf16<<<(32768 + B - 1) / B, B, 0, stream>>>(W1, Wsw, 64);
    gemm1_wmma_bf16<<<N / 16, 128, 0, stream>>>(Xb, Wsw, h1, 512, 64);
  } else {
    gemm1_wmma_f32<<<N / 16, 128, 0, stream>>>(x, W1, h1, 512, 64);
  }

  node1_prep<<<(N + B - 1) / B, B, 0, stream>>>(h1, a_src1, a_dst1,
                                                al1, ar1, m1u, den1, out1, N);
  edge1_max<<<(ET * 8 + B - 1) / B, B, 0, stream>>>(ei, E, ET, al1, ar1, m1u);
  edge1_acc<<<(ET * 8 + B - 1) / B, B, 0, stream>>>(ei, E, ET, al1, ar1, m1u,
                                                    h1, den1, out1);
  node1_finish<<<(N * 64 + B - 1) / B, B, 0, stream>>>(out1, den1, b1, N);

  // Layer 2
  node2_proj<<<(N + B - 1) / B, B, 0, stream>>>(out1, W2, a_src2, a_dst2,
                                                h2, al2, ar2, m2u, den2, out2, N);
  edge2_max<<<(ET + B - 1) / B, B, 0, stream>>>(ei, E, ET, al2, ar2, m2u);
  edge2_acc<<<(ET + B - 1) / B, B, 0, stream>>>(ei, E, ET, al2, ar2, m2u,
                                                h2, den2, out2);
  node2_finish<<<(N + B - 1) / B, B, 0, stream>>>(out2, den2, b2, out, N);
}
